// Decoder_25108378812667
// MI455X (gfx1250) — matrix operand order repaired and checked by reading, not compiled, not run on hardware
//
#include <hip/hip_runtime.h>
#include <hip/hip_bf16.h>

typedef _Float16 f16;
typedef _Float16 half16 __attribute__((ext_vector_type(16)));
typedef float    float8 __attribute__((ext_vector_type(8)));
typedef int      i32x4  __attribute__((ext_vector_type(4)));

#define LAYERS 4
#define DMODEL 128
#define HEADS  8
#define DHCAT  1024      // H * D
#define FFND   512
#define BATCH  256
#define SEQ    168
#define TOK    (BATCH*SEQ)   // 43008, multiple of 64
#define NFUT   24

// CDNA5 async global->LDS path (ASYNCcnt-tracked), guarded so compile never breaks.
// Builtin signature (from clang diagnostic): (int4 AS1*, int4 AS3*, imm, imm)
#if defined(__gfx1250__) && __has_builtin(__builtin_amdgcn_global_load_async_to_lds_b128)
#define HAVE_ASYNC_LDS 1
#define AS1P(p) ((__attribute__((address_space(1))) i32x4*)(p))
#define AS3P(p) ((__attribute__((address_space(3))) i32x4*)(p))
#else
#define HAVE_ASYNC_LDS 0
#endif

__device__ __forceinline__ void async_wait0() {
#if defined(__gfx1250__)
#if __has_builtin(__builtin_amdgcn_s_wait_asynccnt)
    __builtin_amdgcn_s_wait_asynccnt(0);
#else
    asm volatile("s_wait_asynccnt 0" ::: "memory");
#endif
#endif
}

// ---------------------------------------------------------------------------
// f32 -> f16 conversion (weights)
// ---------------------------------------------------------------------------
__global__ void cvt_f32_f16_kernel(const float* __restrict__ src,
                                   f16* __restrict__ dst, int n) {
    int i = blockIdx.x * blockDim.x + threadIdx.x;
    if (i < n) dst[i] = (f16)src[i];
}

// ---------------------------------------------------------------------------
// Input embedding: h = (x @ in_w + in_b) * sqrt(D) + pos_enc
// ---------------------------------------------------------------------------
__global__ void embed_kernel(const float* __restrict__ x,
                             const float* __restrict__ in_w,
                             const float* __restrict__ in_b,
                             float* __restrict__ h32,
                             f16* __restrict__ h16) {
    int idx = blockIdx.x * blockDim.x + threadIdx.x;   // t*128 + d
    if (idx >= TOK * DMODEL) return;
    int t = idx >> 7, d = idx & 127;
    int s = t % SEQ;
    const float* xr = x + (size_t)t * 9;
    float acc = in_b[d];
#pragma unroll
    for (int i = 0; i < 9; ++i) acc += xr[i] * in_w[i * DMODEL + d];
    acc *= 11.313708498984761f;                        // sqrt(128)
    float ang = (float)s * powf(10000.0f, -(float)(2 * (d >> 1)) / 128.0f);
    acc += (d & 1) ? cosf(ang) : sinf(ang);
    h32[idx] = acc;
    h16[idx] = (f16)acc;
}

// ---------------------------------------------------------------------------
// Generic WMMA GEMM: C[M,N] = A[M,K](f16) @ B[K,N](f16) + bias, opt ReLU
// Block: 256 threads (8 waves). Tile: 64(M) x 128(N), K-step 32.
// Wave grid 4x2: each wave owns 16(M) x 64(N) = 4 C fragments.
// Epilogue fully specialized at compile time (no runtime branches).
// ---------------------------------------------------------------------------
template<int RELU, int HAS32, int HAS16>
__global__ __launch_bounds__(256)
void gemm_kernel(const f16* __restrict__ A, const f16* __restrict__ Bm,
                 const float* __restrict__ bias,
                 float* __restrict__ C32, f16* __restrict__ C16,
                 int M, int N, int K) {
    __shared__ __align__(16) f16 As[64 * 40];    // [64 rows][K-step 32, pad->40]
    __shared__ __align__(16) f16 Bst[128 * 40];  // transposed: [128 cols][32 k, pad->40]

    const int tid  = threadIdx.x;
    const int lane = tid & 31, wid = tid >> 5;
    const int n0 = blockIdx.x * 128;
    const int m0 = blockIdx.y * 64;
    const int waveM = (wid >> 1) * 16;
    const int waveN = (wid & 1) * 64;
    const int r  = lane & 15;
    const int hi = lane >> 4;

    float8 zero = {};
    float8 c[4];
#pragma unroll
    for (int j = 0; j < 4; ++j) c[j] = zero;

    const int arow = tid >> 2, acol = (tid & 3) * 8;   // A: 8 halves/thread
    const int brow = tid >> 3, bcol = (tid & 7) * 16;  // B: 16 halves/thread

    union HU { half16 h; uint4 u[2]; };

    for (int kk = 0; kk < K; kk += 32) {
        // fetch B chunk to registers (needs transpose, so manual staging)
        uint4 bv0 = *(const uint4*)(Bm + (size_t)(kk + brow) * N + n0 + bcol);
        uint4 bv1 = *(const uint4*)(Bm + (size_t)(kk + brow) * N + n0 + bcol + 8);
#if !HAVE_ASYNC_LDS
        uint4 av  = *(const uint4*)(A + (size_t)(m0 + arow) * K + kk + acol);
#endif
        // prefetch next K-step tiles into cache (global_prefetch_b8)
        if (kk + 32 < K) {
            __builtin_prefetch(A  + (size_t)(m0 + arow) * K + kk + 32 + acol, 0, 0);
            __builtin_prefetch(Bm + (size_t)(kk + 32 + brow) * N + n0 + bcol, 0, 0);
        }
        __syncthreads();   // previous iteration fully consumed
#if HAVE_ASYNC_LDS
        __builtin_amdgcn_global_load_async_to_lds_b128(
            AS1P(A + (size_t)(m0 + arow) * K + kk + acol),
            AS3P(As + arow * 40 + acol), 0, 0);
#else
        *(uint4*)(As + arow * 40 + acol) = av;
#endif
        const f16* bh0 = (const f16*)&bv0;
        const f16* bh1 = (const f16*)&bv1;
#pragma unroll
        for (int i = 0; i < 8; ++i) Bst[(bcol + i) * 40 + brow]     = bh0[i];
#pragma unroll
        for (int i = 0; i < 8; ++i) Bst[(bcol + 8 + i) * 40 + brow] = bh1[i];
#if HAVE_ASYNC_LDS
        async_wait0();     // async LDS writes visible before barrier
#endif
        __syncthreads();

        // A fragment: 16x32, K(p) = (p<8 ? p : p+8) + hi*8 -> two contiguous b128
        HU fa;
        fa.u[0] = *(const uint4*)(As + (waveM + r) * 40 + hi * 8);
        fa.u[1] = *(const uint4*)(As + (waveM + r) * 40 + 16 + hi * 8);
        // Preload all 4 B fragments, then issue WMMAs back-to-back
        HU fb[4];
#pragma unroll
        for (int j = 0; j < 4; ++j) {
            const int colr = waveN + j * 16 + r;
            fb[j].u[0] = *(const uint4*)(Bst + colr * 40 + hi * 8);
            fb[j].u[1] = *(const uint4*)(Bst + colr * 40 + 16 + hi * 8);
        }
#pragma unroll
        for (int j = 0; j < 4; ++j)
            c[j] = __builtin_amdgcn_wmma_f32_16x16x32_f16(
                       false, fa.h, false, fb[j].h, (short)0, c[j], false, false);
    }

    // epilogue: bias always present; everything compile-time specialized
    float bval[4];
#pragma unroll
    for (int j = 0; j < 4; ++j) bval[j] = bias[n0 + waveN + j * 16 + r];
#pragma unroll
    for (int j = 0; j < 4; ++j) {
        int col = n0 + waveN + j * 16 + r;
#pragma unroll
        for (int i = 0; i < 8; ++i) {
            float v = c[j][i] + bval[j];
            if (RELU) v = fmaxf(v, 0.0f);
            int row = m0 + waveM + i + hi * 8;
            size_t o = (size_t)row * N + col;
            if (HAS32) C32[o] = v;
            if (HAS16) C16[o] = (f16)v;
        }
    }
}

// ---------------------------------------------------------------------------
// Flash attention, one wave per (batch, head, 16-query tile).
// Q/K/V/O layout: [B*S, 1024], head h occupies columns h*128..h*128+127.
// V tile staged to LDS asynchronously, overlapped with the QK^T WMMAs.
// ---------------------------------------------------------------------------
__global__ __launch_bounds__(32)
void attn_kernel(const f16* __restrict__ Q, const f16* __restrict__ Kb,
                 const f16* __restrict__ Vb, f16* __restrict__ O, int masked) {
    __shared__ __align__(16) f16 Vl[32 * 128];  // key block of V
    __shared__ __align__(16) f16 Pl[16 * 32];   // probs (C-layout -> A-layout relay)

    const int lane = threadIdx.x;
    const int r  = lane & 15;
    const int hi = lane >> 4;
    const int q0 = blockIdx.x * 16;
    const int bh = blockIdx.y;
    const int b  = bh >> 3, h = bh & 7;
    const size_t base = ((size_t)b * SEQ) * DHCAT + (size_t)h * DMODEL;

    union HU { half16 h; uint4 u[2]; };

    // Preload Q fragments (4 K-steps over D=128), held in registers throughout
    int qr = q0 + r; if (qr > SEQ - 1) qr = SEQ - 1;
    const f16* qrow = Q + base + (size_t)qr * DHCAT;
    HU qf[4];
#pragma unroll
    for (int ks = 0; ks < 4; ++ks) {
        qf[ks].u[0] = *(const uint4*)(qrow + ks * 32 + hi * 8);
        qf[ks].u[1] = *(const uint4*)(qrow + ks * 32 + 16 + hi * 8);
    }

    float8 zero = {};
    float8 oacc[8];
#pragma unroll
    for (int dt = 0; dt < 8; ++dt) oacc[dt] = zero;
    float mrun[8], lrun[8];
#pragma unroll
    for (int i = 0; i < 8; ++i) { mrun[i] = -1e30f; lrun[i] = 0.0f; }

    int nkb = (SEQ + 31) / 32;                       // 6 key blocks
    if (masked) { int lim = (q0 + 15) / 32 + 1; if (lim < nkb) nkb = lim; }

    const float scale = 0.08838834764831845f;        // 1/sqrt(128)

    for (int kb = 0; kb < nkb; ++kb) {
        const int k0 = kb * 32;

        // Stage V key-block [32 x 128] in LDS (one row per lane).
        // Async path overlaps the copy with the score WMMAs below.
        {
            int vr = k0 + lane; if (vr > SEQ - 1) vr = SEQ - 1;
            const f16* vsrc = Vb + base + (size_t)vr * DHCAT;
            f16* vdst = Vl + lane * 128;
#if HAVE_ASYNC_LDS
#pragma unroll
            for (int j = 0; j < 16; ++j)
                __builtin_amdgcn_global_load_async_to_lds_b128(
                    AS1P(vsrc + j * 8), AS3P(vdst + j * 8), 0, 0);
#else
#pragma unroll
            for (int j = 0; j < 16; ++j)
                ((uint4*)vdst)[j] = ((const uint4*)vsrc)[j];
#endif
        }

        // Scores: [16q x 32k] as two 16x16 WMMA column tiles, contract over D.
        // K fragments load straight from global (row-major K == B-frag layout).
        float8 sc[2];
#pragma unroll
        for (int nt = 0; nt < 2; ++nt) {
            int key = k0 + nt * 16 + r;
            int kc  = key > SEQ - 1 ? SEQ - 1 : key;
            const f16* krow = Kb + base + (size_t)kc * DHCAT;
            HU fb[4];
#pragma unroll
            for (int ks = 0; ks < 4; ++ks) {
                fb[ks].u[0] = *(const uint4*)(krow + ks * 32 + hi * 8);
                fb[ks].u[1] = *(const uint4*)(krow + ks * 32 + 16 + hi * 8);
            }
            float8 cacc = zero;
#pragma unroll
            for (int ks = 0; ks < 4; ++ks)
                cacc = __builtin_amdgcn_wmma_f32_16x16x32_f16(
                           false, qf[ks].h, false, fb[ks].h, (short)0, cacc, false, false);
            sc[nt] = cacc;
        }

        // Online softmax (rows live across 16-lane groups of the C layout)
        float corr[8];
#pragma unroll
        for (int i = 0; i < 8; ++i) {
            int qrow_i = q0 + i + hi * 8;
            float s0 = sc[0][i] * scale;
            float s1 = sc[1][i] * scale;
            int key0 = k0 + r, key1 = k0 + 16 + r;
            if (key0 >= SEQ) s0 = -1e30f; else if (masked && key0 > qrow_i) s0 -= 1e9f;
            if (key1 >= SEQ) s1 = -1e30f; else if (masked && key1 > qrow_i) s1 -= 1e9f;
            float mx = fmaxf(s0, s1);
            mx = fmaxf(mx, __shfl_xor(mx, 1));
            mx = fmaxf(mx, __shfl_xor(mx, 2));
            mx = fmaxf(mx, __shfl_xor(mx, 4));
            mx = fmaxf(mx, __shfl_xor(mx, 8));
            float mnew = fmaxf(mrun[i], mx);
            float p0 = __expf(s0 - mnew);
            float p1 = __expf(s1 - mnew);
            float rs = p0 + p1;
            rs += __shfl_xor(rs, 1);
            rs += __shfl_xor(rs, 2);
            rs += __shfl_xor(rs, 4);
            rs += __shfl_xor(rs, 8);
            float cr = __expf(mrun[i] - mnew);
            lrun[i] = lrun[i] * cr + rs;
            mrun[i] = mnew;
            corr[i] = cr;
            Pl[(i + hi * 8) * 32 + r]      = (f16)p0;
            Pl[(i + hi * 8) * 32 + 16 + r] = (f16)p1;
        }
#pragma unroll
        for (int dt = 0; dt < 8; ++dt) {
#pragma unroll
            for (int i = 0; i < 8; ++i) oacc[dt][i] *= corr[i];
        }

#if HAVE_ASYNC_LDS
        async_wait0();   // V tile resident in LDS before the gather below
#endif
        // Reload P as A fragment (same-wave LDS ops are in-order)
        HU pf;
        pf.u[0] = *(const uint4*)(Pl + r * 32 + hi * 16);
        pf.u[1] = *(const uint4*)(Pl + r * 32 + hi * 16 + 8);

        // O += P @ V  (contract over 32 keys; 8 column tiles over D=128)
#pragma unroll
        for (int dt = 0; dt < 8; ++dt) {
            half16 vb;
#pragma unroll
            for (int p = 0; p < 16; ++p)
                vb[p] = Vl[(p + hi * 16) * 128 + dt * 16 + r];
            oacc[dt] = __builtin_amdgcn_wmma_f32_16x16x32_f16(
                           false, pf.h, false, vb, (short)0, oacc[dt], false, false);
        }
    }

    float inv[8];
#pragma unroll
    for (int i = 0; i < 8; ++i) inv[i] = 1.0f / lrun[i];
#pragma unroll
    for (int i = 0; i < 8; ++i) {
        int q = q0 + i + hi * 8;
        if (q < SEQ) {
            f16* orow = O + base + (size_t)q * DHCAT;
#pragma unroll
            for (int dt = 0; dt < 8; ++dt)
                orow[dt * 16 + r] = (f16)(oacc[dt][i] * inv[i]);
        }
    }
}

// ---------------------------------------------------------------------------
// Residual add + LayerNorm over D=128; one row per 128-thread block
// ---------------------------------------------------------------------------
__global__ __launch_bounds__(128)
void add_ln_kernel(const float* __restrict__ a, const float* __restrict__ res,
                   const float* __restrict__ g, const float* __restrict__ bta,
                   float* __restrict__ out32, f16* __restrict__ out16) {
    __shared__ float red[128];
    int row = blockIdx.x, d = threadIdx.x;
    size_t o = (size_t)row * DMODEL + d;
    float v = a[o] + res[o];
    red[d] = v; __syncthreads();
    for (int st = 64; st > 0; st >>= 1) { if (d < st) red[d] += red[d + st]; __syncthreads(); }
    float mean = red[0] * (1.0f / 128.0f); __syncthreads();
    float c = v - mean;
    red[d] = c * c; __syncthreads();
    for (int st = 64; st > 0; st >>= 1) { if (d < st) red[d] += red[d + st]; __syncthreads(); }
    float var = red[0] * (1.0f / 128.0f);
    float y = c * rsqrtf(var + 1e-9f) * g[d] + bta[d];
    out32[o] = y;
    out16[o] = (f16)y;
}

// ---------------------------------------------------------------------------
// Output head: out[b,j] = h[b, S-24+j, :] . out_w + out_b  (last 24 positions)
// ---------------------------------------------------------------------------
__global__ __launch_bounds__(128)
void head_kernel(const float* __restrict__ h32, const float* __restrict__ out_w,
                 const float* __restrict__ out_b, float* __restrict__ out) {
    __shared__ float red[128];
    int idx = blockIdx.x;            // b*24 + j
    int b = idx / NFUT, j = idx % NFUT;
    int s = SEQ - NFUT + j;
    int d = threadIdx.x;
    float v = h32[((size_t)(b * SEQ + s)) * DMODEL + d] * out_w[d];
    red[d] = v; __syncthreads();
    for (int st = 64; st > 0; st >>= 1) { if (d < st) red[d] += red[d + st]; __syncthreads(); }
    if (d == 0) out[idx] = red[0] + out_b[0];
}

// ---------------------------------------------------------------------------
// Host orchestration
// ---------------------------------------------------------------------------
static void run_gemm(hipStream_t stream, const f16* A, const f16* Bm,
                     const float* bias, float* C32, f16* C16,
                     int M, int N, int K, int relu) {
    dim3 grid(N / 128, M / 64);
    if (C16 && relu)
        gemm_kernel<1, 0, 1><<<grid, 256, 0, stream>>>(A, Bm, bias, C32, C16, M, N, K);
    else if (C16)
        gemm_kernel<0, 0, 1><<<grid, 256, 0, stream>>>(A, Bm, bias, C32, C16, M, N, K);
    else
        gemm_kernel<0, 1, 0><<<grid, 256, 0, stream>>>(A, Bm, bias, C32, C16, M, N, K);
}

extern "C" void kernel_launch(void* const* d_in, const int* in_sizes, int n_in,
                              void* d_out, int out_size, void* d_ws, size_t ws_size,
                              hipStream_t stream) {
    const float* x    = (const float*)d_in[0];
    // d_in[1] = mask (recomputed analytically inside attention kernel)
    const float* in_w = (const float*)d_in[2];
    const float* in_b = (const float*)d_in[3];
    const float* ow   = (const float*)d_in[4];
    const float* ob   = (const float*)d_in[5];
    const float* wq_m = (const float*)d_in[6];  const float* bq_m = (const float*)d_in[7];
    const float* wk_m = (const float*)d_in[8];  const float* bk_m = (const float*)d_in[9];
    const float* wv_m = (const float*)d_in[10]; const float* bv_m = (const float*)d_in[11];
    const float* wo_m = (const float*)d_in[12]; const float* bo_m = (const float*)d_in[13];
    const float* wq_n = (const float*)d_in[14]; const float* bq_n = (const float*)d_in[15];
    const float* wk_n = (const float*)d_in[16]; const float* bk_n = (const float*)d_in[17];
    const float* wv_n = (const float*)d_in[18]; const float* bv_n = (const float*)d_in[19];
    const float* wo_n = (const float*)d_in[20]; const float* bo_n = (const float*)d_in[21];
    const float* w1   = (const float*)d_in[22]; const float* b1   = (const float*)d_in[23];
    const float* w2   = (const float*)d_in[24]; const float* b2   = (const float*)d_in[25];
    const float* ln1g = (const float*)d_in[26]; const float* ln1b = (const float*)d_in[27];
    const float* ln2g = (const float*)d_in[28]; const float* ln2b = (const float*)d_in[29];
    const float* ln3g = (const float*)d_in[30]; const float* ln3b = (const float*)d_in[31];

    char* ws = (char*)d_ws;
    size_t off = 0;
    auto walloc = [&](size_t bytes) -> void* {
        void* p = ws + off;
        off += (bytes + 255) & ~(size_t)255;
        return p;
    };

    const size_t WQKV = (size_t)LAYERS * DMODEL * DHCAT;   // 524288
    const size_t WO   = (size_t)LAYERS * DHCAT * DMODEL;   // 524288
    const size_t W1S  = (size_t)LAYERS * DMODEL * FFND;    // 262144
    const size_t W2S  = (size_t)LAYERS * FFND * DMODEL;    // 262144

    f16* wq_m16 = (f16*)walloc(WQKV * 2);
    f16* wk_m16 = (f16*)walloc(WQKV * 2);
    f16* wv_m16 = (f16*)walloc(WQKV * 2);
    f16* wo_m16 = (f16*)walloc(WO * 2);
    f16* wq_n16 = (f16*)walloc(WQKV * 2);
    f16* wk_n16 = (f16*)walloc(WQKV * 2);
    f16* wv_n16 = (f16*)walloc(WQKV * 2);
    f16* wo_n16 = (f16*)walloc(WO * 2);
    f16* w1_16  = (f16*)walloc(W1S * 2);
    f16* w2_16  = (f16*)walloc(W2S * 2);

    float* h32    = (float*)walloc((size_t)TOK * DMODEL * 4);
    f16*   h16    = (f16*)  walloc((size_t)TOK * DMODEL * 2);
    float* tmp32  = (float*)walloc((size_t)TOK * DMODEL * 4);
    f16*   q16    = (f16*)  walloc((size_t)TOK * DHCAT * 2);
    f16*   k16    = (f16*)  walloc((size_t)TOK * DHCAT * 2);
    f16*   v16    = (f16*)  walloc((size_t)TOK * DHCAT * 2);
    f16*   attn16 = (f16*)  walloc((size_t)TOK * DHCAT * 2);
    f16*   ffn16  = (f16*)  walloc((size_t)TOK * FFND * 2);

    auto cvt = [&](const float* src, f16* dst, size_t n) {
        cvt_f32_f16_kernel<<<(unsigned)((n + 255) / 256), 256, 0, stream>>>(src, dst, (int)n);
    };
    cvt(wq_m, wq_m16, WQKV); cvt(wk_m, wk_m16, WQKV); cvt(wv_m, wv_m16, WQKV);
    cvt(wo_m, wo_m16, WO);
    cvt(wq_n, wq_n16, WQKV); cvt(wk_n, wk_n16, WQKV); cvt(wv_n, wv_n16, WQKV);
    cvt(wo_n, wo_n16, WO);
    cvt(w1, w1_16, W1S); cvt(w2, w2_16, W2S);

    embed_kernel<<<(TOK * DMODEL + 255) / 256, 256, 0, stream>>>(x, in_w, in_b, h32, h16);

    dim3 agrid((SEQ + 15) / 16, BATCH * HEADS);

    for (int l = 0; l < LAYERS; ++l) {
        const size_t wqo = (size_t)l * DMODEL * DHCAT;
        const size_t woo = (size_t)l * DHCAT * DMODEL;

        // ---- masked MHA ----
        run_gemm(stream, h16, wq_m16 + wqo, bq_m + l * DHCAT, nullptr, q16, TOK, DHCAT, DMODEL, 0);
        run_gemm(stream, h16, wk_m16 + wqo, bk_m + l * DHCAT, nullptr, k16, TOK, DHCAT, DMODEL, 0);
        run_gemm(stream, h16, wv_m16 + wqo, bv_m + l * DHCAT, nullptr, v16, TOK, DHCAT, DMODEL, 0);
        attn_kernel<<<agrid, 32, 0, stream>>>(q16, k16, v16, attn16, 1);
        run_gemm(stream, attn16, wo_m16 + woo, bo_m + l * DMODEL, tmp32, nullptr, TOK, DMODEL, DHCAT, 0);
        add_ln_kernel<<<TOK, 128, 0, stream>>>(tmp32, h32, ln1g + l * DMODEL,
                                               ln1b + l * DMODEL, h32, h16);

        // ---- unmasked MHA ----
        run_gemm(stream, h16, wq_n16 + wqo, bq_n + l * DHCAT, nullptr, q16, TOK, DHCAT, DMODEL, 0);
        run_gemm(stream, h16, wk_n16 + wqo, bk_n + l * DHCAT, nullptr, k16, TOK, DHCAT, DMODEL, 0);
        run_gemm(stream, h16, wv_n16 + wqo, bv_n + l * DHCAT, nullptr, v16, TOK, DHCAT, DMODEL, 0);
        attn_kernel<<<agrid, 32, 0, stream>>>(q16, k16, v16, attn16, 0);
        run_gemm(stream, attn16, wo_n16 + woo, bo_n + l * DMODEL, tmp32, nullptr, TOK, DMODEL, DHCAT, 0);
        add_ln_kernel<<<TOK, 128, 0, stream>>>(tmp32, h32, ln2g + l * DMODEL,
                                               ln2b + l * DMODEL, h32, h16);

        // ---- FFN ----
        run_gemm(stream, h16, w1_16 + (size_t)l * DMODEL * FFND, b1 + l * FFND,
                 nullptr, ffn16, TOK, FFND, DMODEL, 1);
        run_gemm(stream, ffn16, w2_16 + (size_t)l * FFND * DMODEL, b2 + l * DMODEL,
                 tmp32, nullptr, TOK, DMODEL, FFND, 0);
        add_ln_kernel<<<TOK, 128, 0, stream>>>(tmp32, h32, ln3g + l * DMODEL,
                                               ln3b + l * DMODEL, h32, h16);
    }

    head_kernel<<<BATCH * NFUT, 128, 0, stream>>>(h32, ow, ob, (float*)d_out);
}
